// GraphSAGE_30614526886306
// MI455X (gfx1250) — matrix operand order repaired and checked by reading, not compiled, not run on hardware
//
#include <hip/hip_runtime.h>

#define N_NODES 40000
#define N_EDGES 640000
#define IN_FEATS 128
#define HIDDEN 256

typedef __attribute__((ext_vector_type(16))) __bf16          v16bf;
typedef __attribute__((ext_vector_type(8)))  float           v8f;
typedef __attribute__((ext_vector_type(8)))  unsigned short  u16x8;

// ---------------- helpers ----------------

__device__ __forceinline__ unsigned short f2bf_rne(float f) {
    unsigned int u = __float_as_uint(f);
    unsigned int r = u + 0x7FFFu + ((u >> 16) & 1u);   // round-to-nearest-even
    return (unsigned short)(r >> 16);
}

// A fragment (16-bit A 16x32): lane holds K = 16*hi .. 16*hi+15, contiguous
__device__ __forceinline__ v16bf load_frag_a(const unsigned short* p) {
    union { u16x8 h[2]; v16bf v; } u;
    u.h[0] = *(const u16x8*)(p);        // K = 16*hi .. 16*hi+7
    u.h[1] = *(const u16x8*)(p + 8);    // K = 16*hi+8 .. 16*hi+15
    return u.v;
}

// B fragment (16-bit B 32x16): lane holds K = 16*hi .. 16*hi+15, contiguous
__device__ __forceinline__ v16bf load_frag_b(const unsigned short* p) {
    union { u16x8 h[2]; v16bf v; } u;
    u.h[0] = *(const u16x8*)(p);        // K = 16*hi .. 16*hi+7
    u.h[1] = *(const u16x8*)(p + 8);    // K = 16*hi+8 .. 16*hi+15
    return u.v;
}

// ---------------- tiny utility kernels ----------------

__global__ void k_zero_f32(float* __restrict__ p, int n) {
    int i = blockIdx.x * blockDim.x + threadIdx.x;
    if (i < n) p[i] = 0.0f;
}

__global__ void k_degree(const int* __restrict__ dst, float* __restrict__ deg) {
    int e = blockIdx.x * blockDim.x + threadIdx.x;
    if (e < N_EDGES) atomicAdd(&deg[dst[e]], 1.0f);
}

__global__ void k_invdeg(const float* __restrict__ deg, float* __restrict__ invd) {
    int i = blockIdx.x * blockDim.x + threadIdx.x;
    if (i < N_NODES) invd[i] = 1.0f / fmaxf(deg[i], 1.0f);
}

// Edge scatter: grid.y indexes the float4 chunk of the feature vector.
__global__ void k_scatter_add(const float* __restrict__ h,
                              const int* __restrict__ src,
                              const int* __restrict__ dst,
                              float* __restrict__ agg, int d) {
    int e = blockIdx.x * blockDim.x + threadIdx.x;
    if (e >= N_EDGES) return;
    int c = blockIdx.y << 2;                  // float offset of this chunk
    const float4 v = *(const float4*)(h + (size_t)src[e] * d + c);
    float* o = agg + (size_t)dst[e] * d + c;
    atomicAdd(o + 0, v.x);
    atomicAdd(o + 1, v.y);
    atomicAdd(o + 2, v.z);
    atomicAdd(o + 3, v.w);
}

// Fused fp32->bf16 convert for both GEMM A operands; folds mean (1/deg) into
// the neighbor activations.
__global__ void k_convert_acts(const float* __restrict__ h,
                               const float* __restrict__ agg,
                               const float* __restrict__ invd,
                               unsigned short* __restrict__ hbf,
                               unsigned short* __restrict__ nbf,
                               int n, int log2d) {
    int i = blockIdx.x * blockDim.x + threadIdx.x;
    if (i >= n) return;
    hbf[i] = f2bf_rne(h[i]);
    nbf[i] = f2bf_rne(agg[i] * invd[i >> log2d]);
}

// W[K][256] fp32 row-major  ->  Wt[256][K] bf16 (so B fragments are contiguous)
__global__ void k_convert_wT(const float* __restrict__ W,
                             unsigned short* __restrict__ Wt, int K) {
    int i = blockIdx.x * blockDim.x + threadIdx.x;
    if (i >= K * HIDDEN) return;
    int k = i >> 8;            // row in W
    int n = i & 255;           // col in W
    Wt[(size_t)n * K + k] = f2bf_rne(W[i]);
}

// ---------------- fused dual-GEMM + bias + ReLU via WMMA ----------------
// out[M,256] = act( hbf[M,K] @ Wself[K,256] + nbf[M,K] @ Wneigh[K,256] + b )
// Block: 256 threads = 8 waves. Wave w owns the 16x16 tile at
// (row0 = 16*blockIdx.x, col0 = 128*blockIdx.y + 16*w).
__global__ void k_sage_gemm(const unsigned short* __restrict__ hbf,
                            const unsigned short* __restrict__ nbf,
                            const unsigned short* __restrict__ wsT,   // [256][K]
                            const unsigned short* __restrict__ wnT,   // [256][K]
                            const float* __restrict__ bias,
                            float* __restrict__ out,
                            int K, int relu) {
    const int lane  = threadIdx.x & 31;
    const int wave  = threadIdx.x >> 5;
    const int l16   = lane & 15;
    const int hi    = lane >> 4;            // 0: lanes 0-15, 1: lanes 16-31
    const int ka    = hi << 4;              // A per-lane K offset (0 or 16)
    const int kb    = hi << 4;              // B per-lane K offset (0 or 16)

    const int row  = blockIdx.x * 16 + l16;             // A row for this lane
    const int ncol = blockIdx.y * 128 + wave * 16 + l16;// B col / out col

    v8f c = {};

    // ---- pass 1: self term ----
    {
        const unsigned short* a = hbf + (size_t)row  * K + ka;
        const unsigned short* b = wsT + (size_t)ncol * K + kb;
        for (int k = 0; k < K; k += 32) {
            __builtin_prefetch(a + k + 32, 0, 1);
            __builtin_prefetch(b + k + 32, 0, 1);
            v16bf af = load_frag_a(a + k);
            v16bf bf = load_frag_b(b + k);
            c = __builtin_amdgcn_wmma_f32_16x16x32_bf16(
                    false, af, false, bf, (short)0, c, false, false);
        }
    }
    // ---- pass 2: neighbor term (accumulates into same C) ----
    {
        const unsigned short* a = nbf + (size_t)row  * K + ka;
        const unsigned short* b = wnT + (size_t)ncol * K + kb;
        for (int k = 0; k < K; k += 32) {
            __builtin_prefetch(a + k + 32, 0, 1);
            __builtin_prefetch(b + k + 32, 0, 1);
            v16bf af = load_frag_a(a + k);
            v16bf bf = load_frag_b(b + k);
            c = __builtin_amdgcn_wmma_f32_16x16x32_bf16(
                    false, af, false, bf, (short)0, c, false, false);
        }
    }

    // ---- epilogue: bias + relu + store (C layout: M = r + 8*hi, N = l16) ----
    const float bv = bias[ncol];
    const int mbase = blockIdx.x * 16 + hi * 8;
#pragma unroll
    for (int r = 0; r < 8; ++r) {
        float v = c[r] + bv;
        if (relu) v = fmaxf(v, 0.0f);
        out[(size_t)(mbase + r) * HIDDEN + ncol] = v;
    }
}

// ---------------- host side ----------------

static inline int ceil_div(int a, int b) { return (a + b - 1) / b; }

struct Ws {
    float*          h32;    // [N_NODES*HIDDEN] fp32 intermediate activations
    float*          agg;    // [N_NODES*HIDDEN] fp32 neighbor sums
    unsigned short* hbf;    // [N_NODES*HIDDEN] bf16 self activations
    unsigned short* nbf;    // [N_NODES*HIDDEN] bf16 mean-neighbor activations
    unsigned short* wsT;    // [HIDDEN*HIDDEN] bf16 W_self^T
    unsigned short* wnT;    // [HIDDEN*HIDDEN] bf16 W_neigh^T
    float*          invd;   // [N_NODES]
    float*          deg;    // [N_NODES]
};

static void run_layer(const Ws& w,
                      const float* h_in, int d,
                      const int* src, const int* dst,
                      const float* Wself, const float* Wneigh, const float* bias,
                      float* out, int relu, hipStream_t stream) {
    const int nelem = N_NODES * d;
    const int log2d = (d == 128) ? 7 : 8;

    k_zero_f32<<<ceil_div(nelem, 256), 256, 0, stream>>>(w.agg, nelem);

    dim3 sgrid(ceil_div(N_EDGES, 256), d / 4);
    k_scatter_add<<<sgrid, 256, 0, stream>>>(h_in, src, dst, w.agg, d);

    k_convert_acts<<<ceil_div(nelem, 256), 256, 0, stream>>>(
        h_in, w.agg, w.invd, w.hbf, w.nbf, nelem, log2d);

    k_convert_wT<<<ceil_div(d * HIDDEN, 256), 256, 0, stream>>>(Wself, w.wsT, d);
    k_convert_wT<<<ceil_div(d * HIDDEN, 256), 256, 0, stream>>>(Wneigh, w.wnT, d);

    dim3 ggrid(N_NODES / 16, HIDDEN / 128);   // 2500 x 2 blocks, 8 waves each
    k_sage_gemm<<<ggrid, 256, 0, stream>>>(w.hbf, w.nbf, w.wsT, w.wnT,
                                           bias, out, d, relu);
}

extern "C" void kernel_launch(void* const* d_in, const int* in_sizes, int n_in,
                              void* d_out, int out_size, void* d_ws, size_t ws_size,
                              hipStream_t stream) {
    const float* feas     = (const float*)d_in[0];
    const int*   src      = (const int*)  d_in[1];
    const int*   dst      = (const int*)  d_in[2];
    const float* W_self1  = (const float*)d_in[3];
    const float* W_neigh1 = (const float*)d_in[4];
    const float* b1       = (const float*)d_in[5];
    const float* W_self2  = (const float*)d_in[6];
    const float* W_neigh2 = (const float*)d_in[7];
    const float* b2       = (const float*)d_in[8];
    const float* W_self3  = (const float*)d_in[9];
    const float* W_neigh3 = (const float*)d_in[10];
    const float* b3       = (const float*)d_in[11];

    const size_t NH = (size_t)N_NODES * HIDDEN;

    Ws w;
    char* p = (char*)d_ws;
    w.h32  = (float*)p;          p += NH * sizeof(float);
    w.agg  = (float*)p;          p += NH * sizeof(float);
    w.hbf  = (unsigned short*)p; p += NH * sizeof(unsigned short);
    w.nbf  = (unsigned short*)p; p += NH * sizeof(unsigned short);
    w.wsT  = (unsigned short*)p; p += (size_t)HIDDEN * HIDDEN * sizeof(unsigned short);
    w.wnT  = (unsigned short*)p; p += (size_t)HIDDEN * HIDDEN * sizeof(unsigned short);
    w.invd = (float*)p;          p += (size_t)N_NODES * sizeof(float);
    w.deg  = (float*)p;          p += (size_t)N_NODES * sizeof(float);

    // degrees (shared by all 3 layers)
    k_zero_f32<<<ceil_div(N_NODES, 256), 256, 0, stream>>>(w.deg, N_NODES);
    k_degree<<<ceil_div(N_EDGES, 256), 256, 0, stream>>>(dst, w.deg);
    k_invdeg<<<ceil_div(N_NODES, 256), 256, 0, stream>>>(w.deg, w.invd);

    // layer 1: feas[40000x128] -> h32 (ReLU)
    run_layer(w, feas, IN_FEATS, src, dst, W_self1, W_neigh1, b1, w.h32, 1, stream);
    // layer 2: h32[40000x256] -> h32 (ReLU)   (GEMM reads only bf16 copies)
    run_layer(w, w.h32, HIDDEN, src, dst, W_self2, W_neigh2, b2, w.h32, 1, stream);
    // layer 3: h32 -> d_out (no ReLU)
    run_layer(w, w.h32, HIDDEN, src, dst, W_self3, W_neigh3, b3, (float*)d_out, 0, stream);
}
